// BitLinear_5660766896531
// MI455X (gfx1250) — 8-bit matrix operand order repaired and checked by reading, compiled, not run on hardware
//
#include <hip/hip_runtime.h>
#include <hip/hip_bf16.h>
#include <stdint.h>


typedef int v8i __attribute__((ext_vector_type(8)));

#define B_ 4
#define S_ 4096
#define M_ (B_ * S_)
#define K_ 2048
#define N_ 2048
#define NW_ ((size_t)N_ * K_)

__global__ void absum_partial_kernel(const float* __restrict__ w,
                                     float* __restrict__ partials) {
    __shared__ float red[256];
    const int t = threadIdx.x;
    float s = 0.0f;
    for (int i = 0; i < 16; ++i) {
        size_t idx = ((size_t)i * 65536u + (size_t)blockIdx.x * 256u + t) * 4u;
        float4 v = *(const float4*)(w + idx);
        s += fabsf(v.x) + fabsf(v.y) + fabsf(v.z) + fabsf(v.w);
    }
    red[t] = s;
    __syncthreads();
    for (int st = 128; st > 0; st >>= 1) {
        if (t < st) red[t] += red[t + st];
        __syncthreads();
    }
    if (t == 0) partials[blockIdx.x] = red[0];
}

__global__ void absum_final_kernel(const float* __restrict__ partials,
                                   float* __restrict__ inv_gamma) {
    __shared__ float red[256];
    const int t = threadIdx.x;
    red[t] = partials[t];
    __syncthreads();
    for (int st = 128; st > 0; st >>= 1) {
        if (t < st) red[t] += red[t + st];
        __syncthreads();
    }
    if (t == 0) {
        float gamma = red[0] * (1.0f / (float)NW_);
        inv_gamma[0] = 1.0f / (gamma + 1e-5f);
    }
}

__global__ void w_quant_kernel(const float* __restrict__ w,
                               const float* __restrict__ inv_gamma,
                               signed char* __restrict__ wq) {
    const float ig = inv_gamma[0];
    size_t i = ((size_t)blockIdx.x * 256u + threadIdx.x) * 4u;
    float4 v = *(const float4*)(w + i);
    union { signed char c[4]; int word; } pk;
    float q0 = fminf(fmaxf(rintf(v.x * ig), -1.0f), 1.0f);
    float q1 = fminf(fmaxf(rintf(v.y * ig), -1.0f), 1.0f);
    float q2 = fminf(fmaxf(rintf(v.z * ig), -1.0f), 1.0f);
    float q3 = fminf(fmaxf(rintf(v.w * ig), -1.0f), 1.0f);
    pk.c[0] = (signed char)q0; pk.c[1] = (signed char)q1;
    pk.c[2] = (signed char)q2; pk.c[3] = (signed char)q3;
    *(int*)(wq + i) = pk.word;
}

__global__ void act_quant_kernel(const float* __restrict__ x,
                                 signed char* __restrict__ qx,
                                 float* __restrict__ rcp_scale) {
    __shared__ float rsum[256];
    __shared__ float rmax[256];
    const int t = threadIdx.x;
    const size_t base = (size_t)blockIdx.x * K_;

    float4 v0 = *(const float4*)(x + base + t * 8);
    float4 v1 = *(const float4*)(x + base + t * 8 + 4);

    float ss = v0.x * v0.x + v0.y * v0.y + v0.z * v0.z + v0.w * v0.w
             + v1.x * v1.x + v1.y * v1.y + v1.z * v1.z + v1.w * v1.w;
    float ma = fmaxf(fmaxf(fmaxf(fabsf(v0.x), fabsf(v0.y)),
                           fmaxf(fabsf(v0.z), fabsf(v0.w))),
                     fmaxf(fmaxf(fabsf(v1.x), fabsf(v1.y)),
                           fmaxf(fabsf(v1.z), fabsf(v1.w))));
    rsum[t] = ss;
    rmax[t] = ma;
    __syncthreads();
    for (int st = 128; st > 0; st >>= 1) {
        if (t < st) {
            rsum[t] += rsum[t + st];
            rmax[t] = fmaxf(rmax[t], rmax[t + st]);
        }
        __syncthreads();
    }
    const float rms    = sqrtf(rsum[0] * (1.0f / (float)K_));
    const float inv_d  = 1.0f / fmaxf(rms, 1e-12f);
    const float mnorm  = rmax[0] * inv_d;
    const float scale  = 127.0f / fmaxf(mnorm, 1e-5f);
    if (t == 0) rcp_scale[blockIdx.x] = fmaxf(mnorm, 1e-5f) * (1.0f / 127.0f);

    const float sq = inv_d * scale;
    union { signed char c[8]; int2 v; } pk;
    float f[8] = { v0.x, v0.y, v0.z, v0.w, v1.x, v1.y, v1.z, v1.w };
#pragma unroll
    for (int i = 0; i < 8; ++i) {
        float q = rintf(f[i] * sq);
        q = fminf(fmaxf(q, -128.0f), 127.0f);
        pk.c[i] = (signed char)q;
    }
    *(int2*)(qx + base + t * 8) = pk.v;
}

#define LDS_STRIDE 80

__global__ void __launch_bounds__(256)
bitlinear_gemm_kernel(const signed char* __restrict__ qx,
                      const signed char* __restrict__ wq,
                      const float* __restrict__ rcp_scale,
                      float* __restrict__ out) {
    __shared__ __align__(16) unsigned char ldsA[128 * LDS_STRIDE];
    __shared__ __align__(16) unsigned char ldsB[128 * LDS_STRIDE];

    const int tid    = threadIdx.x;
    const int lane   = tid & 31;
    const int wid    = tid >> 5;
    const int wave_m = wid >> 2;
    const int wave_n = wid & 3;
    const int lm     = lane & 15;
    const int hi     = lane >> 4;

    const int bm = blockIdx.x * 128;
    const int bn = blockIdx.y * 128;

    v8i acc[4][2];
#pragma unroll
    for (int ms = 0; ms < 4; ++ms)
#pragma unroll
        for (int ns = 0; ns < 2; ++ns) acc[ms][ns] = (v8i)(0);

    for (int k0 = 0; k0 < K_; k0 += 64) {
#pragma unroll
        for (int i = 0; i < 2; ++i) {
            int chunk = tid + i * 256;
            int row   = chunk >> 2;
            int seg   = (chunk & 3) * 16;
            const signed char* ga = qx + (size_t)(bm + row) * K_ + k0 + seg;
            const signed char* gb = wq + (size_t)(bn + row) * K_ + k0 + seg;
            *(int4*)(ldsA + row * LDS_STRIDE + seg) = *(const int4*)ga;
            *(int4*)(ldsB + row * LDS_STRIDE + seg) = *(const int4*)gb;
            if (k0 + 64 < K_) {
                __builtin_prefetch(ga + 64, 0, 0);
                __builtin_prefetch(gb + 64, 0, 0);
            }
        }
        __syncthreads();

        v8i bfrag[2];
#pragma unroll
        for (int ns = 0; ns < 2; ++ns) {
            int col = wave_n * 32 + ns * 16 + lm;
            const unsigned char* p = ldsB + col * LDS_STRIDE + hi * 16;
            union { v8i v; int4 q[2]; } b;
            b.q[0] = *(const int4*)(p);
            b.q[1] = *(const int4*)(p + 32);
            bfrag[ns] = b.v;
        }

#pragma unroll
        for (int ms = 0; ms < 4; ++ms) {
            int row = wave_m * 64 + ms * 16 + lm;
            const unsigned char* pa = ldsA + row * LDS_STRIDE + hi * 16;
            union { v8i v; int4 q[2]; } a;
            a.q[0] = *(const int4*)(pa);
            a.q[1] = *(const int4*)(pa + 32);
#pragma unroll
            for (int ns = 0; ns < 2; ++ns) {
                acc[ms][ns] = __builtin_amdgcn_wmma_i32_16x16x64_iu8(
                     true, a.v,  true, bfrag[ns],
                    acc[ms][ns],  false,  false);
            }
        }
        __syncthreads();
    }

#pragma unroll
    for (int ms = 0; ms < 4; ++ms) {
        int row0 = bm + wave_m * 64 + ms * 16 + hi * 8;
#pragma unroll
        for (int ns = 0; ns < 2; ++ns) {
            int col = bn + wave_n * 32 + ns * 16 + lm;
#pragma unroll
            for (int r = 0; r < 8; ++r) {
                int m = row0 + r;
                out[(size_t)m * N_ + col] = (float)acc[ms][ns][r] * rcp_scale[m];
            }
        }
    }
}

extern "C" void kernel_launch(void* const* d_in, const int* in_sizes, int n_in,
                              void* d_out, int out_size, void* d_ws, size_t ws_size,
                              hipStream_t stream) {
    const float* x = (const float*)d_in[0];
    const float* w = (const float*)d_in[1];
    float* out = (float*)d_out;

    unsigned char* ws = (unsigned char*)d_ws;
    signed char* qx  = (signed char*)ws;
    signed char* wq  = (signed char*)(ws + (size_t)M_ * K_);
    float* rcp       = (float*)(ws + (size_t)M_ * K_ + NW_);
    float* partials  = rcp + M_;
    float* inv_gamma = partials + 256;

    absum_partial_kernel<<<256, 256, 0, stream>>>(w, partials);
    absum_final_kernel<<<1, 256, 0, stream>>>(partials, inv_gamma);

    w_quant_kernel<<<(int)(NW_ / (256 * 4)), 256, 0, stream>>>(w, inv_gamma, wq);

    act_quant_kernel<<<M_, 256, 0, stream>>>(x, qx, rcp);

    dim3 grid(M_ / 128, N_ / 128);
    bitlinear_gemm_kernel<<<grid, 256, 0, stream>>>(qx, wq, rcp, out);
}
